// KipfMLPGNN_65085934403743
// MI455X (gfx1250) — matrix operand order repaired and checked by reading, not compiled, not run on hardware
//
#include <hip/hip_runtime.h>
#include <hip/hip_bf16.h>

// ---------------------------------------------------------------------------
// Problem constants (match the JAX reference)
// ---------------------------------------------------------------------------
#define BB   32              // batch
#define NN   64              // nodes
#define DD   128             // embedding dim
#define HH   256             // hidden dim
#define LL   3               // layers
#define ROWS (BB * NN)       // 2048 embedding rows
#define HPAD 264             // padded LDS row stride (bf16 elements) for bank spread

typedef __bf16  bf16_t;
typedef __attribute__((ext_vector_type(16))) __bf16 v16bf;
typedef __attribute__((ext_vector_type(8)))  __bf16 v8bf;
typedef __attribute__((ext_vector_type(4)))  __bf16 v4bf;
typedef __attribute__((ext_vector_type(8)))  float  v8f;
typedef __attribute__((ext_vector_type(4)))  float  v4f;

__device__ __forceinline__ v16bf load_frag(const bf16_t* lo, const bf16_t* hi) {
    v8bf a = *(const v8bf*)lo;
    v8bf b = *(const v8bf*)hi;
    return __builtin_shufflevector(a, b, 0,1,2,3,4,5,6,7,8,9,10,11,12,13,14,15);
}

__device__ __forceinline__ float relu_f(float x) { return fmaxf(x, 0.0f); }

// ---------------------------------------------------------------------------
// Kernel 0: embedding gather + bf16 mirror.  E[row] = table[x[row]]
// ---------------------------------------------------------------------------
__global__ __launch_bounds__(DD)
void gather_kernel(const int* __restrict__ x, const float* __restrict__ table,
                   float* __restrict__ E, bf16_t* __restrict__ Ebf) {
    const int row = blockIdx.x;
    const int d   = threadIdx.x;
    const int tok = x[row];
    const float v = table[tok * DD + d];
    E  [row * DD + d] = v;
    Ebf[row * DD + d] = (bf16_t)v;
}

// ---------------------------------------------------------------------------
// Kernel 0b: weight re-pack + f32->bf16 convert (done once).
//   Wc (1024 x 128): rows 0..255   = W1_0[:, :128]   (sender half, MLP0)
//                    rows 256..511 = W1_0[:, 128:]   (receiver half, MLP0)
//                    rows 512..767 = W1_1[:, :128]
//                    rows 768..1023= W1_1[:, 128:]
//   W2bf{0,1} (128 x 256): bf16 copies of W2_{0,1}
// ---------------------------------------------------------------------------
__global__ __launch_bounds__(256)
void wconv_kernel(const float* __restrict__ W1_0, const float* __restrict__ W1_1,
                  const float* __restrict__ W2_0, const float* __restrict__ W2_1,
                  bf16_t* __restrict__ Wc, bf16_t* __restrict__ W2bf0,
                  bf16_t* __restrict__ W2bf1) {
    const int total = 1024 * DD + 2 * DD * HH;
    const int stride = gridDim.x * blockDim.x;
    for (int i = blockIdx.x * blockDim.x + threadIdx.x; i < total; i += stride) {
        if (i < 1024 * DD) {
            const int n = i >> 7, k = i & (DD - 1);
            float v;
            if      (n <  256) v = W1_0[ n        * 2*DD +       k];
            else if (n <  512) v = W1_0[(n - 256) * 2*DD + DD +  k];
            else if (n <  768) v = W1_1[(n - 512) * 2*DD +       k];
            else               v = W1_1[(n - 768) * 2*DD + DD +  k];
            Wc[i] = (bf16_t)v;
        } else if (i < 1024 * DD + DD * HH) {
            const int j = i - 1024 * DD;
            W2bf0[j] = (bf16_t)W2_0[j];
        } else {
            const int j = i - 1024 * DD - DD * HH;
            W2bf1[j] = (bf16_t)W2_1[j];
        }
    }
}

// ---------------------------------------------------------------------------
// Kernel 1: projection GEMM   P(2048 x 1024) = Ebf(2048 x 128) * Wc^T
// Layout of P columns: [S0 | R0 | S1 | R1], each 256 wide.
// Grid: 32 M-tiles x 8 N-tiles = 256 blocks, 8 waves each, 16 WMMA/wave.
// ---------------------------------------------------------------------------
__global__ __launch_bounds__(256)
void proj_kernel(const bf16_t* __restrict__ Ebf, const bf16_t* __restrict__ Wc,
                 float* __restrict__ P) {
    const int bm   = blockIdx.x >> 3;
    const int bn   = blockIdx.x & 7;
    const int w    = threadIdx.x >> 5;
    const int lane = threadIdx.x & 31;

    const int mbase = bm * 64 + (w & 3) * 16;
    const int nbase = bn * 128 + (w >> 2) * 64;
    const int mrow  = mbase + (lane & 15);
    const int koff  = (lane >> 4) * 8;   // A-frag half-K select (ISA 16-bit A layout)
    const int kb2   = (lane >> 4) * 8;   // B-frag half-K select

    v8f acc[4];
#pragma unroll
    for (int t = 0; t < 4; ++t) acc[t] = (v8f){0,0,0,0,0,0,0,0};

#pragma unroll
    for (int kk = 0; kk < DD; kk += 32) {
        const bf16_t* abase = Ebf + mrow * DD + kk + koff;
        v16bf afrag = load_frag(abase, abase + 16);
        v16bf bfrag[4];
#pragma unroll
        for (int nt = 0; nt < 4; ++nt) {
            const int n = nbase + nt * 16 + (lane & 15);
            const bf16_t* bbase = Wc + n * DD + kk + kb2;
            bfrag[nt] = load_frag(bbase, bbase + 16);
        }
#pragma unroll
        for (int nt = 0; nt < 4; ++nt)
            acc[nt] = __builtin_amdgcn_wmma_f32_16x16x32_bf16(
                false, afrag, false, bfrag[nt], (short)0, acc[nt], false, false);
    }

    // C/D layout: VGPR i of lane L -> (M = i + 8*(L>>4), N = L&15)
    const int mout = mbase + 8 * (lane >> 4);
#pragma unroll
    for (int nt = 0; nt < 4; ++nt) {
        const int n = nbase + nt * 16 + (lane & 15);
#pragma unroll
        for (int i = 0; i < 8; ++i)
            P[(mout + i) * 1024 + n] = acc[nt][i];
    }
}

// ---------------------------------------------------------------------------
// Kernel 2: per-(b,r) edge MLP layer-2 GEMM + arc mix + scatter-add + residual.
//   h{0,1}[c,h] = relu(S{0,1}[b,c,h] + R{0,1}[b,r,h] + b1_{0,1}[h])   (LDS, bf16)
//   m{0,1}     = h{0,1} @ W2_{0,1}^T           (64x256x128 WMMA GEMM)
//   agg[d]     = sum_c relu(m0+b2_0)*(1-a_c) + sum_c relu(m1+b2_1)*a_c
//   (the arc mix is separable, so the two MLPs run as two sequential,
//    software-pipelined GEMM passes that each scatter into agg)
//   E_out[b,r] = E_in[b,r] + agg
// last==0: one workgroup per (b,r), 2048 blocks; updates E (ping-pong).
// last==1: output slice only needs r==0 -> 32 blocks (b, r=0), writes d_out.
// ---------------------------------------------------------------------------
__global__ __launch_bounds__(256)
void edge_kernel(const float* __restrict__ P,
                 const float* __restrict__ b1_0, const float* __restrict__ b1_1,
                 const bf16_t* __restrict__ W2bf0, const bf16_t* __restrict__ W2bf1,
                 const float* __restrict__ b2_0, const float* __restrict__ b2_1,
                 const int* __restrict__ arcs,
                 const float* __restrict__ E_in, float* __restrict__ E_out,
                 bf16_t* __restrict__ Ebf_out, float* __restrict__ out,
                 int last) {
    extern __shared__ __align__(16) char smem[];
    bf16_t* h0  = (bf16_t*)smem;                               // 64 x HPAD bf16
    bf16_t* h1  = (bf16_t*)(smem + NN * HPAD * 2);             // 64 x HPAD bf16
    float*  r0  = (float*)(smem + 2 * NN * HPAD * 2);          // 256
    float*  r1  = r0 + HH;                                     // 256
    float*  aw  = r1 + HH;                                     // 64
    float*  agg = aw + NN;                                     // 128

    const int b    = last ? (int)blockIdx.x : (int)(blockIdx.x >> 6);
    const int r    = last ? 0               : (int)(blockIdx.x & (NN - 1));
    const int rowr = b * NN + r;
    const int tid  = threadIdx.x;

    if (tid == 0) { __builtin_prefetch(W2bf0, 0, 1); __builtin_prefetch(W2bf1, 0, 1); }

    // Phase 0: receiver projections + biases, arc weights, zero agg.
    r0[tid] = P[rowr * 1024 + 256 + tid] + b1_0[tid];
    r1[tid] = P[rowr * 1024 + 768 + tid] + b1_1[tid];
    if (tid < NN) aw[tid] = (float)arcs[(b * NN + r) * NN + tid];
    if (tid < DD) agg[tid] = 0.0f;
    __syncthreads();

    // Phase 1: build h0/h1 in LDS (bf16), vectorized 4-wide.
    for (int idx = tid * 4; idx < NN * HH; idx += 256 * 4) {
        const int c = idx >> 8;
        const int h = idx & (HH - 1);
        v4f s0 = *(const v4f*)&P[(b * NN + c) * 1024 +       h];
        v4f s1 = *(const v4f*)&P[(b * NN + c) * 1024 + 512 + h];
        v4bf p0, p1;
#pragma unroll
        for (int j = 0; j < 4; ++j) {
            p0[j] = (bf16_t)relu_f(s0[j] + r0[h + j]);
            p1[j] = (bf16_t)relu_f(s1[j] + r1[h + j]);
        }
        *(v4bf*)&h0[c * HPAD + h] = p0;
        *(v4bf*)&h1[c * HPAD + h] = p1;
    }
    __syncthreads();

    // Phase 2: two software-pipelined WMMA GEMM passes (MLP0, then MLP1).
    // Wave w owns M rows [16*(w&3)..+15], N cols [64*(w>>2)..+63].
    const int w     = tid >> 5;
    const int lane  = tid & 31;
    const int mbase = (w & 3) * 16;
    const int nbase = (w >> 2) * 64;
    const int arow  = mbase + (lane & 15);
    const int koff  = (lane >> 4) * 8;
    const int kb2   = (lane >> 4) * 8;

    for (int mlp = 0; mlp < 2; ++mlp) {
        const bf16_t* __restrict__ hb = mlp ? h1    : h0;
        const bf16_t* __restrict__ wb = mlp ? W2bf1 : W2bf0;
        const float*  __restrict__ b2 = mlp ? b2_1  : b2_0;

        v8f acc[4];
#pragma unroll
        for (int t = 0; t < 4; ++t) acc[t] = (v8f){0,0,0,0,0,0,0,0};

        const bf16_t* hrow = hb + arow * HPAD + koff;   // + kk at use site

        // Prologue: fragments for k-step 0.
        v16bf aCur = load_frag(hrow, hrow + 16);
        v16bf bCur[4];
#pragma unroll
        for (int nt = 0; nt < 4; ++nt) {
            const bf16_t* p = wb + (nbase + nt * 16 + (lane & 15)) * HH + kb2;
            bCur[nt] = load_frag(p, p + 16);
        }

#pragma unroll
        for (int kk = 0; kk < HH; kk += 32) {
            // Prefetch next k-step while current WMMAs run.
            v16bf aNxt = aCur;
            v16bf bNxt[4];
#pragma unroll
            for (int t = 0; t < 4; ++t) bNxt[t] = bCur[t];
            if (kk + 32 < HH) {
                const bf16_t* ha = hrow + kk + 32;
                aNxt = load_frag(ha, ha + 16);           // ds_load_b128 x2
#pragma unroll
                for (int nt = 0; nt < 4; ++nt) {
                    const bf16_t* p = wb + (nbase + nt * 16 + (lane & 15)) * HH
                                    + kk + 32 + kb2;
                    bNxt[nt] = load_frag(p, p + 16);     // global_load_b128 x2
                }
            }
#pragma unroll
            for (int nt = 0; nt < 4; ++nt)
                acc[nt] = __builtin_amdgcn_wmma_f32_16x16x32_bf16(
                    false, aCur, false, bCur[nt], (short)0, acc[nt], false, false);
            aCur = aNxt;
#pragma unroll
            for (int t = 0; t < 4; ++t) bCur[t] = bNxt[t];
        }

        // Reduce over M (= sender index c) with separable arc weight.
#pragma unroll
        for (int nt = 0; nt < 4; ++nt) {
            const int d = nbase + nt * 16 + (lane & 15);
            const float bias = b2[d];
            float partial = 0.0f;
#pragma unroll
            for (int i = 0; i < 8; ++i) {
                const int c = mbase + 8 * (lane >> 4) + i;
                const float wgt = mlp ? aw[c] : (1.0f - aw[c]);
                partial += relu_f(acc[nt][i] + bias) * wgt;
            }
            atomicAdd(&agg[d], partial);   // ds_add_f32
        }
    }
    __syncthreads();

    // Phase 3: residual update.  Last layer writes only the output slice.
    if (tid < DD) {
        const float val = E_in[rowr * DD + tid] + agg[tid];
        if (last) {
            out[b * DD + tid] = val;
        } else {
            E_out [rowr * DD + tid] = val;
            Ebf_out[rowr * DD + tid] = (bf16_t)val;
        }
    }
}

// ---------------------------------------------------------------------------
// Host-side orchestration
// ---------------------------------------------------------------------------
extern "C" void kernel_launch(void* const* d_in, const int* in_sizes, int n_in,
                              void* d_out, int out_size, void* d_ws, size_t ws_size,
                              hipStream_t stream) {
    (void)in_sizes; (void)n_in; (void)out_size; (void)ws_size;

    const int*   x      = (const int*)  d_in[0];
    const int*   arcs   = (const int*)  d_in[1];
    const float* table  = (const float*)d_in[3];
    const float* W1_0   = (const float*)d_in[4];
    const float* b1_0   = (const float*)d_in[5];
    const float* W2_0   = (const float*)d_in[6];
    const float* b2_0   = (const float*)d_in[7];
    const float* W1_1   = (const float*)d_in[8];
    const float* b1_1   = (const float*)d_in[9];
    const float* W2_1   = (const float*)d_in[10];
    const float* b2_1   = (const float*)d_in[11];
    float* out = (float*)d_out;

    char* ws = (char*)d_ws;
    size_t off = 0;
    float*  Ea    = (float*) (ws + off); off += (size_t)ROWS * DD * 4;
    float*  Eb    = (float*) (ws + off); off += (size_t)ROWS * DD * 4;
    bf16_t* Ebfa  = (bf16_t*)(ws + off); off += (size_t)ROWS * DD * 2;
    bf16_t* Ebfb  = (bf16_t*)(ws + off); off += (size_t)ROWS * DD * 2;
    float*  P     = (float*) (ws + off); off += (size_t)ROWS * 1024 * 4;
    bf16_t* Wc    = (bf16_t*)(ws + off); off += (size_t)1024 * DD * 2;
    bf16_t* W2bf0 = (bf16_t*)(ws + off); off += (size_t)DD * HH * 2;
    bf16_t* W2bf1 = (bf16_t*)(ws + off); off += (size_t)DD * HH * 2;

    gather_kernel<<<ROWS, DD, 0, stream>>>(x, table, Ea, Ebfa);
    wconv_kernel<<<256, 256, 0, stream>>>(W1_0, W1_1, W2_0, W2_1, Wc, W2bf0, W2bf1);

    const size_t smem = (size_t)(2 * NN * HPAD * 2)            // h0 + h1 (bf16)
                      + (size_t)(2 * HH + NN + DD) * 4;        // r0, r1, aw, agg
    for (int layer = 0; layer < LL; ++layer) {
        const float*  E_in    = (layer & 1) ? Eb   : Ea;
        float*        E_out   = (layer & 1) ? Ea   : Eb;
        const bf16_t* Ebf_in  = (layer & 1) ? Ebfb : Ebfa;
        bf16_t*       Ebf_out = (layer & 1) ? Ebfa : Ebfb;
        const int last = (layer == LL - 1) ? 1 : 0;
        proj_kernel<<<256, 256, 0, stream>>>(Ebf_in, Wc, P);
        edge_kernel<<<last ? BB : ROWS, 256, smem, stream>>>(
            P, b1_0, b1_1, W2bf0, W2bf1, b2_0, b2_1, arcs, E_in, E_out,
            Ebf_out, out, last);
    }
}
